// AttentionPooling_46651934769555
// MI455X (gfx1250) — compile-verified, matrix operand order checked by reading, not run on hardware
//
#include <hip/hip_runtime.h>
#include <math.h>

// ---------------------------------------------------------------------------
// Types for WMMA (CDNA5 / gfx1250, wave32)
// ---------------------------------------------------------------------------
typedef __attribute__((ext_vector_type(16))) __bf16 v16bf;
typedef __attribute__((ext_vector_type(8)))  float  v8f;
typedef __attribute__((ext_vector_type(8)))  unsigned int v8u;

#define D_IN   256
#define D_HID  128

// Truncating pack of two f32 -> one dword of two bf16 via a single v_perm_b32:
// result bytes = {hi[3], hi[2], lo[3], lo[2]}
__device__ __forceinline__ unsigned pk2bf(float lo, float hi) {
    return __builtin_amdgcn_perm(__float_as_uint(hi), __float_as_uint(lo),
                                 0x07060302u);
}

// gelu(tanh approx) == v * sigmoid(2*sqrt(2/pi)*(v + 0.044715 v^3)),
// built from native v_exp_f32 / v_rcp_f32 (no libm special-case bloat).
__device__ __forceinline__ float geluf(float v) {
    const float k1   = 0.044715f;
    const float m2k0 = -1.5957691216057308f;      // -2*sqrt(2/pi)
    float u = m2k0 * fmaf(k1 * v * v, v, v);      // -2*k0*(v + k1*v^3)
    float e = __expf(u);
    return v * __builtin_amdgcn_rcpf(1.0f + e);   // v * sigmoid
}

__device__ __forceinline__ int lower_bound_i32(const int* b, int n, int key) {
    int lo = 0, hi = n;
    while (lo < hi) {
        int mid = (lo + hi) >> 1;
        if (b[mid] < key) lo = mid + 1; else hi = mid;
    }
    return lo;
}

// ---------------------------------------------------------------------------
// Kernel 0: pack W1 (f32 row-major [K=256][N=128]) into bf16 arranged in the
// exact wave32 B-fragment order: index = ((kt*8 + t)*32 + lane)*16 + j
//   kOff = (lane<16)?0:8 ; k = kt*32 + kOff + (j&7) + (j>=8 ? 16 : 0)
//   n = t*16 + (lane&15)
// so the logits kernel loads one contiguous, coalesced 32B v8u per lane.
// ---------------------------------------------------------------------------
__global__ void pack_w1_kernel(const float* __restrict__ W1,
                               unsigned short* __restrict__ w1p) {
    int idx = blockIdx.x * blockDim.x + threadIdx.x;
    if (idx >= 8 * 8 * 32 * 16) return;
    int j    = idx & 15;
    int lane = (idx >> 4) & 31;
    int t    = (idx >> 9) & 7;
    int kt   = idx >> 12;
    int kOff = (lane < 16) ? 0 : 8;
    int k    = kt * 32 + kOff + (j & 7) + ((j >= 8) ? 16 : 0);
    int n    = t * 16 + (lane & 15);
    unsigned u = __float_as_uint(W1[k * D_HID + n]);
    w1p[idx] = (unsigned short)((u + 0x7FFFu + ((u >> 16) & 1u)) >> 16);
}

// ---------------------------------------------------------------------------
// Kernel 1: logits[i] = gelu(x[i] @ W1 + b1) @ W2 + b2
// One wave per 16-row tile. K-loop kept rolled (unroll 1) so live set is
// acc(64) + A(8) + B(8) + temps -> stays under 256 VGPRs (no vgpr_msb churn).
// ---------------------------------------------------------------------------
__global__ __launch_bounds__(256) void logits_kernel(
        const float* __restrict__ x,
        const unsigned int* __restrict__ w1p,   // packed bf16 B fragments
        const float* __restrict__ b1,
        const float* __restrict__ W2,
        const float* __restrict__ b2,
        float* __restrict__ logits,
        int N) {
    int wave = (blockIdx.x * blockDim.x + threadIdx.x) >> 5;
    int lane = threadIdx.x & 31;
    int nTiles = (N + 15) >> 4;
    if (wave >= nTiles) return;

    int rowBase = wave << 4;
    int row = rowBase + (lane & 15);
    if (row >= N) row = N - 1;               // clamp loads for the tail tile
    const int kOff = (lane < 16) ? 0 : 8;
    const float* xr = x + (size_t)row * D_IN + kOff;
    const v8u* wp = (const v8u*)w1p + lane;

    v8f acc[8];
    #pragma unroll
    for (int t = 0; t < 8; ++t)
        #pragma unroll
        for (int v = 0; v < 8; ++v) acc[t][v] = 0.0f;

    #pragma unroll 1
    for (int kt = 0; kt < 8; ++kt) {
        // ---- build bf16 A fragment (16x32) from f32 x -------------------
        const float4* p = (const float4*)(xr + kt * 32);
        float4 p0 = p[0], p1 = p[1], q0 = p[4], q1 = p[5];  // +0,+4,+16,+20
        v8u ap;
        ap[0] = pk2bf(p0.x, p0.y); ap[1] = pk2bf(p0.z, p0.w);
        ap[2] = pk2bf(p1.x, p1.y); ap[3] = pk2bf(p1.z, p1.w);
        ap[4] = pk2bf(q0.x, q0.y); ap[5] = pk2bf(q0.z, q0.w);
        ap[6] = pk2bf(q1.x, q1.y); ap[7] = pk2bf(q1.z, q1.w);
        v16bf afrag = __builtin_bit_cast(v16bf, ap);

        // ---- 8 N-tiles of W1 -------------------------------------------
        #pragma unroll
        for (int t = 0; t < 8; ++t) {
            v16bf bfrag = __builtin_bit_cast(v16bf, wp[(kt * 8 + t) * 32]);
            acc[t] = __builtin_amdgcn_wmma_f32_16x16x32_bf16(
                false, afrag, false, bfrag, (short)0, acc[t], false, false);
        }
    }

    // ---- epilogue: bias + gelu + dot with W2, reduce over 128 columns ---
    // C layout: VGPR v, lanes 0-15 -> (M=v,   N=lane),
    //                   lanes16-31 -> (M=v+8, N=lane-16)
    float r[8];
    #pragma unroll
    for (int v = 0; v < 8; ++v) r[v] = 0.0f;
    #pragma unroll
    for (int t = 0; t < 8; ++t) {
        int col = t * 16 + (lane & 15);
        float b1v = b1[col];
        float w2v = W2[col];
        #pragma unroll
        for (int v = 0; v < 8; ++v)
            r[v] = fmaf(geluf(acc[t][v] + b1v), w2v, r[v]);
    }
    // reduce across the 16 lanes of each half-wave
    #pragma unroll
    for (int v = 0; v < 8; ++v) {
        float s = r[v];
        s += __shfl_down(s, 8, 16);
        s += __shfl_down(s, 4, 16);
        s += __shfl_down(s, 2, 16);
        s += __shfl_down(s, 1, 16);
        r[v] = s;
    }
    float b2v = b2[0];
    if (lane == 0) {
        #pragma unroll
        for (int v = 0; v < 8; ++v)
            if (rowBase + v < N) logits[rowBase + v] = r[v] + b2v;
    } else if (lane == 16) {
        #pragma unroll
        for (int v = 0; v < 8; ++v)
            if (rowBase + 8 + v < N) logits[rowBase + 8 + v] = r[v] + b2v;
    }
}

// ---------------------------------------------------------------------------
// Kernel 2: per-segment max and sum-of-exp (one wave per segment, batch sorted)
// ---------------------------------------------------------------------------
__global__ __launch_bounds__(256) void segstats_kernel(
        const float* __restrict__ logits,
        const int* __restrict__ batch,
        float* __restrict__ segmax,
        float* __restrict__ segsum,
        int N, int G) {
    int g    = (blockIdx.x * blockDim.x + threadIdx.x) >> 5;
    int lane = threadIdx.x & 31;
    if (g >= G) return;
    int start = lower_bound_i32(batch, N, g);
    int end   = lower_bound_i32(batch, N, g + 1);

    float m = -INFINITY;
    for (int i = start + lane; i < end; i += 32) m = fmaxf(m, logits[i]);
    #pragma unroll
    for (int off = 16; off; off >>= 1) m = fmaxf(m, __shfl_xor(m, off));

    float s = 0.0f;
    for (int i = start + lane; i < end; i += 32) s += __expf(logits[i] - m);
    #pragma unroll
    for (int off = 16; off; off >>= 1) s += __shfl_xor(s, off);

    if (lane == 0) { segmax[g] = m; segsum[g] = s; }
}

// ---------------------------------------------------------------------------
// Kernel 3: attn[i] = exp(logits[i] - segmax[b]) / segsum[b]
// ---------------------------------------------------------------------------
__global__ void attn_kernel(const float* __restrict__ logits,
                            const int* __restrict__ batch,
                            const float* __restrict__ segmax,
                            const float* __restrict__ segsum,
                            float* __restrict__ attn, int N) {
    int i = blockIdx.x * blockDim.x + threadIdx.x;
    if (i >= N) return;
    int g = batch[i];
    attn[i] = __expf(logits[i] - segmax[g]) *
              __builtin_amdgcn_rcpf(segsum[g]);
}

// ---------------------------------------------------------------------------
// Kernel 4: pooled[g,:] = sum_{rows of segment g} x[row,:] * attn[row]
// One block per segment; thread = column. Atomic-free, writes all outputs.
// Row loop unrolled x4 -> 4 independent FMA chains to hide HBM latency.
// ---------------------------------------------------------------------------
__global__ __launch_bounds__(256) void pool_kernel(
        const float* __restrict__ x,
        const float* __restrict__ attn,
        const int* __restrict__ batch,
        float* __restrict__ out, int N) {
    int g = blockIdx.x;
    int d = threadIdx.x;
    int start = lower_bound_i32(batch, N, g);
    int end   = lower_bound_i32(batch, N, g + 1);
    const float* xp = x + d;
    float a0 = 0.0f, a1 = 0.0f, a2 = 0.0f, a3 = 0.0f;
    int r = start;
    for (; r + 3 < end; r += 4) {
        float w0 = attn[r + 0], w1 = attn[r + 1];
        float w2 = attn[r + 2], w3 = attn[r + 3];
        a0 = fmaf(xp[(size_t)(r + 0) * D_IN], w0, a0);
        a1 = fmaf(xp[(size_t)(r + 1) * D_IN], w1, a1);
        a2 = fmaf(xp[(size_t)(r + 2) * D_IN], w2, a2);
        a3 = fmaf(xp[(size_t)(r + 3) * D_IN], w3, a3);
    }
    for (; r < end; ++r)
        a0 = fmaf(xp[(size_t)r * D_IN], attn[r], a0);
    out[(size_t)g * D_IN + d] = (a0 + a1) + (a2 + a3);
}

// ---------------------------------------------------------------------------
// Host-side launcher
// ---------------------------------------------------------------------------
static inline size_t align_up(size_t v, size_t a) { return (v + a - 1) & ~(a - 1); }

extern "C" void kernel_launch(void* const* d_in, const int* in_sizes, int n_in,
                              void* d_out, int out_size, void* d_ws, size_t ws_size,
                              hipStream_t stream) {
    const float* x   = (const float*)d_in[0];   // (N, 256)
    const float* W1  = (const float*)d_in[1];   // (256, 128)
    const float* b1  = (const float*)d_in[2];   // (128,)
    const float* W2  = (const float*)d_in[3];   // (128, 1)
    const float* b2  = (const float*)d_in[4];   // (1,)
    const int*   bat = (const int*)d_in[5];     // (N,) sorted segment ids
    float* out = (float*)d_out;                 // (G, 256)

    const int N = in_sizes[5];
    const int G = out_size / D_IN;

    // workspace carve-out
    char* ws = (char*)d_ws;
    size_t o = 0;
    float* logits = (float*)(ws + o); o = align_up(o + (size_t)N * 4, 256);
    float* attn   = (float*)(ws + o); o = align_up(o + (size_t)N * 4, 256);
    float* segmax = (float*)(ws + o); o = align_up(o + (size_t)G * 4, 256);
    float* segsum = (float*)(ws + o); o = align_up(o + (size_t)G * 4, 256);
    unsigned short* w1p = (unsigned short*)(ws + o);
    o = align_up(o + (size_t)D_IN * D_HID * 2, 256);
    (void)ws_size; (void)n_in;

    // 0) pack W1 -> bf16 B-fragment layout (32768 elems)
    pack_w1_kernel<<<(32768 + 255) / 256, 256, 0, stream>>>(W1, w1p);

    // 1) WMMA logits: one wave per 16 rows
    int waves   = (N + 15) / 16;
    int blocksA = (waves + 7) / 8;
    logits_kernel<<<blocksA, 256, 0, stream>>>(x, (const unsigned int*)w1p,
                                               b1, W2, b2, logits, N);

    // 2) per-segment softmax stats (one wave per segment)
    segstats_kernel<<<(G + 7) / 8, 256, 0, stream>>>(logits, bat, segmax, segsum, N, G);

    // 3) attention weights
    attn_kernel<<<(N + 255) / 256, 256, 0, stream>>>(logits, bat, segmax, segsum, attn, N);

    // 4) weighted pooling, one block per segment
    pool_kernel<<<G, 256, 0, stream>>>(x, attn, bat, out, N);
}
